// ConvolutionalSelfAttention_90632399880598
// MI455X (gfx1250) — matrix operand order repaired and checked by reading, not compiled, not run on hardware
//
#include <hip/hip_runtime.h>

#define HW_  1024
#define C_   128
#define F_   900
#define B_   8
#define EPS_ 1e-5f

typedef __attribute__((ext_vector_type(16))) _Float16 v16h;
typedef __attribute__((ext_vector_type(8)))  _Float16 v8h;
typedef __attribute__((ext_vector_type(8)))  float    v8f;

__device__ __forceinline__ v8f wmma16(v16h a, v16h b, v8f c) {
  // D = A(16x32 f16) * B(32x16 f16) + C(16x16 f32)
  return __builtin_amdgcn_wmma_f32_16x16x32_f16(false, a, false, b, (short)0, c,
                                                false, false);
}

// A-matrix 16x32 f16 fragment (ISA 7.12.2): lane m = lane&15 holds row m,
// khalf = lane>>4; elements 0..7 -> K = k0 + khalf*8 + e, elements 8..15 ->
// K = k0 + 16 + khalf*8 + (e-8).  `row` points at this lane's row.
__device__ __forceinline__ v16h load_a_frag(const _Float16* __restrict__ row,
                                            int lane, int k0) {
  int kh = (lane >> 4) & 1;
  const _Float16* p = row + k0 + kh * 8;
  v8h lo = *(const v8h*)p;
  v8h hi = *(const v8h*)(p + 16);
  return __builtin_shufflevector(lo, hi, 0, 1, 2, 3, 4, 5, 6, 7,
                                 8, 9, 10, 11, 12, 13, 14, 15);
}

// B-matrix 32x16 f16 fragment: lane n = lane&15 holds column n, khalf = lane>>4;
// elements 0..7 -> K = k0 + khalf*8 + e, elements 8..15 -> K = k0 + 16 + khalf*8 + (e-8).
// `row` points at this lane's column
// source row (B stored as [n][k], i.e. we multiply by its transpose).
__device__ __forceinline__ v16h load_b_frag(const _Float16* row, int lane, int k0) {
  int kh = (lane >> 4) & 1;
  const _Float16* p = row + k0 + kh * 8;
  v8h lo = *(const v8h*)p;
  v8h hi = *(const v8h*)(p + 16);
  return __builtin_shufflevector(lo, hi, 0, 1, 2, 3, 4, 5, 6, 7,
                                 8, 9, 10, 11, 12, 13, 14, 15);
}

// Async copy of 16 bytes global->LDS (GV addressing form); tracked by ASYNCcnt.
__device__ __forceinline__ void async_stage_b128(unsigned lds_off, const void* g) {
  asm volatile("global_load_async_to_lds_b128 %0, %1, off"
               :: "v"(lds_off), "v"((unsigned long long)(uintptr_t)g)
               : "memory");
}

// ---------------------------------------------------------------- converts
__global__ void cvt_f32_to_f16(const float* __restrict__ src,
                               _Float16* __restrict__ dst, int n) {
  int i = blockIdx.x * blockDim.x + threadIdx.x;
  if (i < n) dst[i] = (_Float16)src[i];
}

// ---------------------------------------------------------------- V = x @ wv + bv
__global__ void v_proj_kernel(const float* __restrict__ batch,
                              const float* __restrict__ wv,
                              const float* __restrict__ bv,
                              float* __restrict__ Vv) {
  int i = blockIdx.x * blockDim.x + threadIdx.x;  // 0 .. B*HW-1
  if (i >= B_ * HW_) return;
  const float* row = batch + (size_t)i * C_;
  float s = bv[0];
#pragma unroll 4
  for (int c = 0; c < C_; c += 4) {
    float4 x = *(const float4*)(row + c);
    float4 w = *(const float4*)(wv + c);
    s += x.x * w.x + x.y * w.y + x.z * w.z + x.w * w.w;
  }
  Vv[i] = s;
}

// ---------------------------------------------------------------- Q/K projections
// One wave per 16-row i-tile; loops the 8 column tiles; 4 K-chunks each.
// Q and K accumulator chains are interleaved -> independent WMMAs back-to-back.
__global__ void proj_kernel(const _Float16* __restrict__ xh,
                            const _Float16* __restrict__ wqh,
                            const _Float16* __restrict__ wkh,
                            const float* __restrict__ bq,
                            const float* __restrict__ bk,
                            _Float16* __restrict__ Qh,
                            _Float16* __restrict__ Kh) {
  int wave = threadIdx.x >> 5, lane = threadIdx.x & 31;
  int wid = blockIdx.x * 8 + wave;          // 0 .. B*64-1
  int b = wid >> 6, it = wid & 63;
  const _Float16* xb = xh + (size_t)b * HW_ * C_;
  const _Float16* arow = xb + (size_t)(it * 16 + (lane & 15)) * C_;
  v16h a[4];
#pragma unroll
  for (int kc = 0; kc < 4; ++kc) a[kc] = load_a_frag(arow, lane, 32 * kc);

  int n = lane & 15;
  int half = (lane >> 4) & 1;
  _Float16* Qb = Qh + (size_t)b * HW_ * C_;
  _Float16* Kb = Kh + (size_t)b * HW_ * C_;

  for (int nt = 0; nt < 8; ++nt) {
    const _Float16* bqrow = wqh + (size_t)(nt * 16 + n) * C_;
    const _Float16* bkrow = wkh + (size_t)(nt * 16 + n) * C_;
    v8f cq = {}, ck = {};
#pragma unroll
    for (int kc = 0; kc < 4; ++kc) {
      cq = wmma16(a[kc], load_b_frag(bqrow, lane, 32 * kc), cq);
      ck = wmma16(a[kc], load_b_frag(bkrow, lane, 32 * kc), ck);
    }
    float biasq = bq[nt * 16 + n];
    float biask = bk[nt * 16 + n];
    int col = nt * 16 + n;
#pragma unroll
    for (int g = 0; g < 8; ++g) {
      int mm = it * 16 + g + half * 8;
      Qb[(size_t)mm * C_ + col] = (_Float16)(cq[g] + biasq);
      Kb[(size_t)mm * C_ + col] = (_Float16)(ck[g] + biask);
    }
  }
}

// ---------------------------------------------------------------- R/T row stats
// Block = 8 waves, one 16-row i-tile each, all of the same batch. All waves
// consume the SAME 64 Kh q-panels, so each 4KB panel is staged once per block
// into LDS with global_load_async_to_lds_b128 (double-buffered, ASYNCcnt),
// and B-fragments are read back with ds_load_b128. Two independent WMMA
// accumulator chains per tile hide the D->C hazard.
__global__ void rowstats_kernel(const _Float16* __restrict__ Qh,
                                const _Float16* __restrict__ Kh,
                                const float* __restrict__ Vv,
                                float* __restrict__ Rv,
                                float* __restrict__ Tv) {
  __shared__ __align__(32) _Float16 kpanel[2][16 * C_];   // 2 x 4KB
  int wave = threadIdx.x >> 5, lane = threadIdx.x & 31;
  int tid = threadIdx.x;
  int wid = blockIdx.x * 8 + wave;          // 0 .. B*64-1
  int b = wid >> 6, it = wid & 63;
  const _Float16* Qb = Qh + (size_t)b * HW_ * C_;
  const _Float16* Kb = Kh + (size_t)b * HW_ * C_;
  const float* Vb = Vv + (size_t)b * HW_;

  const _Float16* arow = Qb + (size_t)(it * 16 + (lane & 15)) * C_;
  v16h a[4];
#pragma unroll
  for (int kc = 0; kc < 4; ++kc) a[kc] = load_a_frag(arow, lane, 32 * kc);

  float rloc[8], tloc[8];
#pragma unroll
  for (int g = 0; g < 8; ++g) { rloc[g] = 0.f; tloc[g] = 0.f; }

  int n = lane & 15;
  // stage panel 0 (each of the 256 threads moves 16B; +1 ASYNCcnt per wave)
  async_stage_b128((unsigned)(uintptr_t)&kpanel[0][tid * 8], Kb + tid * 8);

  for (int qt = 0; qt < 64; ++qt) {
    if (qt < 63) {
      async_stage_b128((unsigned)(uintptr_t)&kpanel[(qt + 1) & 1][tid * 8],
                       Kb + (size_t)(qt + 1) * 16 * C_ + tid * 8);
      asm volatile("s_wait_asynccnt 0x1" ::: "memory");  // panel qt complete
    } else {
      asm volatile("s_wait_asynccnt 0x0" ::: "memory");
    }
    __syncthreads();                         // all waves' portions visible

    const _Float16* brow = &kpanel[qt & 1][(lane & 15) * C_];
    v16h b0 = load_b_frag(brow, lane, 0);
    v16h b1 = load_b_frag(brow, lane, 32);
    v16h b2 = load_b_frag(brow, lane, 64);
    v16h b3 = load_b_frag(brow, lane, 96);
    v8f c0 = {}, c1 = {};
    c0 = wmma16(a[0], b0, c0);
    c1 = wmma16(a[1], b1, c1);
    c0 = wmma16(a[2], b2, c0);
    c1 = wmma16(a[3], b3, c1);

    float vq = Vb[qt * 16 + n];
#pragma unroll
    for (int g = 0; g < 8; ++g) {
      float e = __expf(c0[g] + c1[g]);
      rloc[g] += e;
      tloc[g] += e * vq;
    }
    __syncthreads();                         // done reading before overwrite
  }
  // sum over N (16 lanes per half-wave); xor of bits 0..3 stays in the group
#pragma unroll
  for (int off = 1; off < 16; off <<= 1) {
#pragma unroll
    for (int g = 0; g < 8; ++g) {
      rloc[g] += __shfl_xor(rloc[g], off, 32);
      tloc[g] += __shfl_xor(tloc[g], off, 32);
    }
  }
  if ((lane & 15) == 0) {
    int mbase = it * 16 + ((lane >> 4) ? 8 : 0);
#pragma unroll
    for (int g = 0; g < 8; ++g) {
      Rv[(size_t)b * HW_ + mbase + g] = rloc[g];
      Tv[(size_t)b * HW_ + mbase + g] = tloc[g];
    }
  }
}

// ---------------------------------------------------------------- per-window output
// One wave per (b,f): recompute the 9x9 local-E block with one padded 16x16
// WMMA tile (two independent accumulator chains), form gates by subtracting
// window contributions from R/T, then out[c] = sum_p gate[p] * x[idx[f,p], c].
__global__ void window_kernel(const float* __restrict__ batch,
                              const _Float16* __restrict__ Qh,
                              const _Float16* __restrict__ Kh,
                              const float* __restrict__ Vv,
                              const float* __restrict__ Rv,
                              const float* __restrict__ Tv,
                              const int* __restrict__ idx,
                              float* __restrict__ out) {
  int wave = threadIdx.x >> 5, lane = threadIdx.x & 31;
  int wid = blockIdx.x * 8 + wave;          // 0 .. B*F-1
  int b = wid / F_, f = wid % F_;
  __shared__ float s_gate[8][12];
  __shared__ int   s_ipos[8][12];

  const _Float16* Qb = Qh + (size_t)b * HW_ * C_;
  const _Float16* Kb = Kh + (size_t)b * HW_ * C_;
  const float* xb = batch + (size_t)b * HW_ * C_;

  int m = lane & 15;
  bool valid = (m < 9);
  int im = idx[f * 9 + (valid ? m : 8)];
  const _Float16* arow = Qb + (size_t)im * C_;
  const _Float16* brow = Kb + (size_t)im * C_;
  // warm the gather rows for the epilogue (global_prefetch_b8)
  __builtin_prefetch(xb + (size_t)im * C_ + ((lane >> 4) ? 64 : 0), 0, 1);
  v16h zero = {};

  v8f acc0 = {}, acc1 = {};
#pragma unroll
  for (int kc = 0; kc < 4; ++kc) {
    v16h a  = load_a_frag(arow, lane, 32 * kc);
    v16h bb = load_b_frag(brow, lane, 32 * kc);
    a  = valid ? a  : zero;   // zero-pad rows m>=9
    bb = valid ? bb : zero;   // zero-pad cols n>=9
    if (kc & 1) acc1 = wmma16(a, bb, acc1);
    else        acc0 = wmma16(a, bb, acc0);
  }

  float vq = valid ? Vv[(size_t)b * HW_ + im] : 0.0f;
  float sE[8], sEV[8];
#pragma unroll
  for (int g = 0; g < 8; ++g) {
    float e = valid ? __expf(acc0[g] + acc1[g]) : 0.0f;  // invalid cols -> 0
    sE[g] = e;
    sEV[g] = e * vq;
  }
#pragma unroll
  for (int off = 1; off < 16; off <<= 1) {
#pragma unroll
    for (int g = 0; g < 8; ++g) {
      sE[g]  += __shfl_xor(sE[g],  off, 32);
      sEV[g] += __shfl_xor(sEV[g], off, 32);
    }
  }
  if ((lane & 15) == 0) {
    int half = (lane >> 4) & 1;
#pragma unroll
    for (int g = 0; g < 8; ++g) {
      int mm = g + half * 8;
      if (mm < 9) {
        int i_m = idx[f * 9 + mm];
        float numer = Tv[(size_t)b * HW_ + i_m] - sEV[g];
        float denom = Rv[(size_t)b * HW_ + i_m] - sE[g] + EPS_;
        s_gate[wave][mm] = numer / denom;
        s_ipos[wave][mm] = i_m;
      }
    }
  }
  __syncthreads();

  int c0 = lane * 4;                          // 32 lanes * 4 = 128 channels
  float o0 = 0.f, o1 = 0.f, o2 = 0.f, o3 = 0.f;
#pragma unroll
  for (int p = 0; p < 9; ++p) {
    float g = s_gate[wave][p];
    float4 x = *(const float4*)(xb + (size_t)s_ipos[wave][p] * C_ + c0);
    o0 += g * x.x; o1 += g * x.y; o2 += g * x.z; o3 += g * x.w;
  }
  float4 res; res.x = o0; res.y = o1; res.z = o2; res.w = o3;
  *(float4*)(out + (size_t)wid * C_ + c0) = res;
}

// ---------------------------------------------------------------- launch
extern "C" void kernel_launch(void* const* d_in, const int* in_sizes, int n_in,
                              void* d_out, int out_size, void* d_ws, size_t ws_size,
                              hipStream_t stream) {
  const float* batch = (const float*)d_in[0];   // (B,32,32,128) f32
  const float* Wq    = (const float*)d_in[1];   // (128,128)
  const float* bq    = (const float*)d_in[2];   // (128,)
  const float* Wk    = (const float*)d_in[3];   // (128,128)
  const float* bk    = (const float*)d_in[4];   // (128,)
  const float* wv    = (const float*)d_in[5];   // (128,1)
  const float* bv    = (const float*)d_in[6];   // (1,)
  // d_in[7] = gmask (unused: handled algebraically)
  const int*   idx   = (const int*)d_in[8];     // (900,9) int32
  float* out = (float*)d_out;                   // (B,30,30,128) f32

  char* ws = (char*)d_ws;
  _Float16* xh  = (_Float16*)(ws + 0);          // 2 MB
  _Float16* Qh  = (_Float16*)(ws + (2u << 20)); // 2 MB
  _Float16* Kh  = (_Float16*)(ws + (4u << 20)); // 2 MB
  _Float16* wqh = (_Float16*)(ws + (6u << 20));          // 32 KB
  _Float16* wkh = (_Float16*)(ws + (6u << 20) + 32768);  // 32 KB
  float*    Vv  = (float*)(ws + (6u << 20) + 65536);     // 32 KB
  float*    Rv  = (float*)(ws + (6u << 20) + 98304);     // 32 KB
  float*    Tv  = (float*)(ws + (6u << 20) + 131072);    // 32 KB

  int nbig = B_ * HW_ * C_;                     // 1,048,576
  cvt_f32_to_f16<<<(nbig + 255) / 256, 256, 0, stream>>>(batch, xh, nbig);
  cvt_f32_to_f16<<<64, 256, 0, stream>>>(Wq, wqh, C_ * C_);
  cvt_f32_to_f16<<<64, 256, 0, stream>>>(Wk, wkh, C_ * C_);
  v_proj_kernel<<<(B_ * HW_) / 256, 256, 0, stream>>>(batch, wv, bv, Vv);

  proj_kernel<<<B_ * 64 / 8, 256, 0, stream>>>(xh, wqh, wkh, bq, bk, Qh, Kh);
  rowstats_kernel<<<B_ * 64 / 8, 256, 0, stream>>>(Qh, Kh, Vv, Rv, Tv);
  window_kernel<<<B_ * F_ / 8, 256, 0, stream>>>(batch, Qh, Kh, Vv, Rv, Tv, idx, out);
}
